// LSTM_82506321756677
// MI455X (gfx1250) — matrix operand order repaired and checked by reading, not compiled, not run on hardware
//
#include <hip/hip_runtime.h>
#include <hip/hip_fp16.h>

// ---------------------------------------------------------------------------
// LayerNorm-LSTM for MI455X (gfx1250, wave32, WMMA + TDM).
// S=1024, B=32, I=H=512, L=2, 4H=2048.
// ---------------------------------------------------------------------------

typedef _Float16 v16h __attribute__((ext_vector_type(16)));
typedef float    v8f  __attribute__((ext_vector_type(8)));
typedef unsigned int u32x4 __attribute__((ext_vector_type(4)));
typedef int          i32x4 __attribute__((ext_vector_type(4)));
typedef int          i32x8 __attribute__((ext_vector_type(8)));

#if defined(__has_builtin)
#if __has_builtin(__builtin_amdgcn_tensor_load_to_lds)
#define HAVE_TDM 1
#endif
#endif

#define EPS 1e-5f
#define S_LEN 1024
#define BATCH 32
#define IDIM  512
#define HDIM  512
#define NG    2048              // 4*H
#define NT_G  (NG / 16)         // 128 N-tiles
#define KT_X  (IDIM / 32)       // 16 K-tiles (K=512, k-step 32)
#define MROWS (S_LEN * BATCH)   // 32768
#define NWG_REC 16
#define REC_THREADS 128

#define WMMA_F16(a, b, c)                                                  \
  __builtin_amdgcn_wmma_f32_16x16x32_f16(false, (a), false, (b), (short)0, \
                                         (c), false, false)

// ---------------------------------------------------------------------------
// Weight pack: f32 [K,N] row-major -> f16 B-fragments.
// B 32x16 f16 layout: lane = (n%16) + 16*((k%32>>3)&1), e = (k%8) | ((k%32>>4)<<3),
// fragment = ((k/32)*NT + n/16)*512 + lane*16 + e  (32B contiguous per lane).
// ---------------------------------------------------------------------------
__global__ void pack_b_kernel(const float* __restrict__ src,
                              _Float16* __restrict__ dst, int K, int N) {
  int idx = blockIdx.x * 256 + threadIdx.x;
  if (idx >= K * N) return;
  int k = idx / N, n = idx % N;
  int kt = k >> 5, kl = k & 31;
  int lane = (n & 15) + (((kl >> 3) & 1) << 4);
  int e = (kl & 7) | ((kl >> 4) << 3);
  dst[((size_t)kt * (N >> 4) + (n >> 4)) * 512 + lane * 16 + e] =
      (_Float16)src[idx];
}

// ---------------------------------------------------------------------------
// Activation pack: f32 [M,K] row-major -> f16 A-fragments.
// A 16x32 f16 layout (ISA 7.12.2): lane = (m%16) + 16*hi, hi = (k%32>>3)&1,
// e = (k%8) | ((k%32>>4)<<3), fragment = ((m/16)*(K/32) + k/32)*512 + lane*16+e.
// ---------------------------------------------------------------------------
__global__ void pack_a_kernel(const float* __restrict__ src,
                              _Float16* __restrict__ dst, int M, int K) {
  int idx = blockIdx.x * 256 + threadIdx.x;
  if (idx >= M * K) return;
  int m = idx / K, k = idx % K;
  int mt = m >> 4, kt = k >> 5, kl = k & 31;
  int hi = (kl >> 3) & 1;
  int e = (kl & 7) | ((kl >> 4) << 3);
  int lane = (m & 15) + (hi << 4);
  dst[((size_t)mt * (K >> 5) + kt) * 512 + lane * 16 + e] = (_Float16)src[idx];
}

// ---------------------------------------------------------------------------
// Big GEMM: gx_raw[M,NG] = Apk x Bpk, f16 in / f32 accumulate.
// WG = 4 waves, 64x64 tile; wave = 16(M) x 64(N) -> 4 accumulators.
// Register double-buffered: next k-iteration's 5 fragments are in flight
// while 4 WMMAs consume the current ones (hides load latency).
// ---------------------------------------------------------------------------
__global__ __launch_bounds__(128) void gemm_wmma_kernel(
    const _Float16* __restrict__ Apk, const _Float16* __restrict__ Bpk,
    float* __restrict__ out) {
  const int lane = threadIdx.x & 31;
  const int wave = threadIdx.x >> 5;
  const int mt = blockIdx.y * 4 + wave;

  const _Float16* aPtr = Apk + (size_t)mt * KT_X * 512 + lane * 16;
  const _Float16* bPtr = Bpk + (size_t)blockIdx.x * 4 * 512 + lane * 16;

  v8f acc[4] = {};
  v16h a = *(const v16h*)aPtr;
  v16h b0 = *(const v16h*)(bPtr + 0 * 512);
  v16h b1 = *(const v16h*)(bPtr + 1 * 512);
  v16h b2 = *(const v16h*)(bPtr + 2 * 512);
  v16h b3 = *(const v16h*)(bPtr + 3 * 512);

  for (int kt = 0; kt < KT_X - 1; ++kt) {
    const _Float16* bp = bPtr + (size_t)(kt + 1) * NT_G * 512;
    v16h an = *(const v16h*)(aPtr + (kt + 1) * 512);
    v16h c0 = *(const v16h*)(bp + 0 * 512);
    v16h c1 = *(const v16h*)(bp + 1 * 512);
    v16h c2 = *(const v16h*)(bp + 2 * 512);
    v16h c3 = *(const v16h*)(bp + 3 * 512);
    acc[0] = WMMA_F16(a, b0, acc[0]);
    acc[1] = WMMA_F16(a, b1, acc[1]);
    acc[2] = WMMA_F16(a, b2, acc[2]);
    acc[3] = WMMA_F16(a, b3, acc[3]);
    a = an; b0 = c0; b1 = c1; b2 = c2; b3 = c3;
  }
  acc[0] = WMMA_F16(a, b0, acc[0]);
  acc[1] = WMMA_F16(a, b1, acc[1]);
  acc[2] = WMMA_F16(a, b2, acc[2]);
  acc[3] = WMMA_F16(a, b3, acc[3]);

  // C/D layout: lane -> n = lane%16, VGPR r -> m = r + 8*(lane>=16).
  const int m0 = mt * 16 + ((lane >> 4) << 3);
  const int n0 = blockIdx.x * 64 + (lane & 15);
#pragma unroll
  for (int nt = 0; nt < 4; ++nt)
#pragma unroll
    for (int r = 0; r < 8; ++r)
      out[(size_t)(m0 + r) * NG + n0 + nt * 16] = acc[nt][r];
}

// ---------------------------------------------------------------------------
// Row LayerNorm over NG=2048 (one wave per row), fused with lnA affine and
// the recurrent bias fold:  gx = LN(raw)*w + b + bias_l   (in place).
// ---------------------------------------------------------------------------
__global__ __launch_bounds__(128) void ln_rows_kernel(
    float* __restrict__ gx, const float* __restrict__ lnw,
    const float* __restrict__ lnb, const float* __restrict__ bias) {
  const int lane = threadIdx.x & 31;
  const int wave = threadIdx.x >> 5;
  float* p = gx + (size_t)(blockIdx.x * 4 + wave) * NG;
  float v[64], s = 0.f, sq = 0.f;
#pragma unroll
  for (int i = 0; i < 64; ++i) {
    v[i] = p[lane + i * 32];
    s += v[i];
    sq += v[i] * v[i];
  }
#pragma unroll
  for (int m = 16; m >= 1; m >>= 1) {
    s += __shfl_xor(s, m, 32);
    sq += __shfl_xor(sq, m, 32);
  }
  float mean = s * (1.f / NG);
  float var = sq * (1.f / NG) - mean * mean;
  float rs = __frsqrt_rn(var + EPS);
#pragma unroll
  for (int i = 0; i < 64; ++i) {
    int n = lane + i * 32;
    p[n] = (v[i] - mean) * rs * lnw[n] + lnb[n] + bias[n];
  }
}

__device__ __forceinline__ float fsig(float x) {
  return 1.f / (1.f + __expf(-x));
}

__device__ __forceinline__ void grid_sync(unsigned* c) {
  __syncthreads();
  if (threadIdx.x == 0) {
    __hip_atomic_fetch_add(c, 1u, __ATOMIC_ACQ_REL, __HIP_MEMORY_SCOPE_AGENT);
    while (__hip_atomic_load(c, __ATOMIC_ACQUIRE, __HIP_MEMORY_SCOPE_AGENT) <
           NWG_REC)
      __builtin_amdgcn_s_sleep(1);
  }
  __syncthreads();
}

// ---------------------------------------------------------------------------
// Persistent recurrent scan. 16 WGs x 4 waves. WG w owns 32 hidden units
// j in [w*32, w*32+32) -> gate N-tiles NT(g,t) = g*32 + w*2 + t (g=0..3).
// Wave = (mtile, t); its 4 accumulators are the 4 gates of the same hidden
// units -> i/f/o/u math is register local. wh streams from L2 (f16 packed,
// double-buffered fragments), h is DMA'd into LDS each step by the Tensor
// Data Mover. 3 grid barriers/step (two LN reductions + h republish).
// ---------------------------------------------------------------------------
__global__ __launch_bounds__(REC_THREADS) void lstm_scan_kernel(
    const float* __restrict__ gx, const _Float16* __restrict__ whpk,
    _Float16* __restrict__ hA, const float* __restrict__ c0,
    const float* __restrict__ lnh_w, const float* __restrict__ lnh_b,
    const float* __restrict__ lnc_w, const float* __restrict__ lnc_b,
    float* __restrict__ xout, float* __restrict__ hfin,
    float* __restrict__ cfin, float* __restrict__ gstats,
    float* __restrict__ cstats, unsigned* __restrict__ bars) {
  __shared__ _Float16 hs[2 * 16 * 512];  // 32KB: packed A-fragments of h

  const int tid = threadIdx.x;
  const int lane = tid & 31;
  const int wave = tid >> 5;
  const int w = blockIdx.x;
  const int mtile = wave & 1;
  const int t = wave >> 1;
  const int nlo = lane & 15;
  const int hi = lane >> 4;
  const int j = w * 32 + t * 16 + nlo;  // hidden unit column

  float creg[8];
#pragma unroll
  for (int r = 0; r < 8; ++r)
    creg[r] = c0[(mtile * 16 + r + 8 * hi) * HDIM + j];

  float lnhw[4], lnhb[4];
#pragma unroll
  for (int g = 0; g < 4; ++g) {
    int n = (g * 32 + w * 2 + t) * 16 + nlo;
    lnhw[g] = lnh_w[n];
    lnhb[g] = lnh_b[n];
  }
  const float lncw = lnc_w[j], lncb = lnc_b[j];

#if defined(HAVE_TDM)
  // Loop-invariant TDM descriptor: 1-row 2D tile of 16384 x 2-byte elements
  // (the whole packed-h block), global hA -> LDS hs.
  const unsigned lds_off = (unsigned)(size_t)(&hs[0]);
  const unsigned long long ga = (unsigned long long)(size_t)hA;
  u32x4 g0;
  g0[0] = 1u;                                           // count=1 (valid D#)
  g0[1] = lds_off;                                      // lds_addr
  g0[2] = (unsigned)(ga & 0xffffffffu);                 // global_addr[31:0]
  g0[3] = (unsigned)((ga >> 32) & 0x01ffffffu) | (2u << 30);  // addr + type=2
  i32x8 g1;
  g1[0] = (int)(1u << 16);       // wg_mask=0, data_size=1 (2 bytes)
  g1[1] = (int)(16384u << 16);   // tensor_dim0[15:0] in bits[63:48]
  g1[2] = (int)(1u << 16);       // tensor_dim0 hi=0, tensor_dim1=1
  g1[3] = (int)(16384u << 16);   // tensor_dim1 hi=0, tile_dim0=16384
  g1[4] = 1;                     // tile_dim1=1, tile_dim2=0
  g1[5] = 16384;                 // tensor_dim0_stride low
  g1[6] = 0;
  g1[7] = 0;
  const i32x4 z4 = {0, 0, 0, 0};
#if defined(__clang_major__) && (__clang_major__ >= 23)
  const i32x8 z8 = {0, 0, 0, 0, 0, 0, 0, 0};
#endif
#endif

  // wh fragment base for this WG's gate strips: nt(g) = g*32 + w*2 + t.
  const _Float16* bBase = whpk + (size_t)(w * 2 + t) * 512 + lane * 16;

  for (int s = 0; s < S_LEN; ++s) {
    // ---- stage packed h into LDS (TDM DMA; fallback: manual b128 copy) ----
#if defined(HAVE_TDM)
    if (tid < 32) {  // one TDM op per WG (wave 0, full EXEC)
#if defined(__clang_major__) && (__clang_major__ >= 23)
      __builtin_amdgcn_tensor_load_to_lds(g0, g1, z4, z4, z8, 0);
#else
      __builtin_amdgcn_tensor_load_to_lds(g0, g1, z4, z4, 0);
#endif
      __builtin_amdgcn_s_wait_tensorcnt(0);
    }
#else
    {
      const uint4* srcv = (const uint4*)hA;
      uint4* dstv = (uint4*)hs;
      for (int i = tid; i < 2048; i += REC_THREADS) dstv[i] = srcv[i];
    }
#endif
    if (s + 1 < S_LEN)
      __builtin_prefetch(&gx[(size_t)((s + 1) * BATCH) * NG + tid * 64], 0, 1);
    __syncthreads();

    // ---- gate GEMM slice: [32 x 128] = h[32x512] x wh_slice ----
    // B fragments double-buffered from L2; A fragments from LDS.
    v8f acc[4] = {};
    v16h b0 = *(const v16h*)(bBase + (size_t)0 * 512);
    v16h b1 = *(const v16h*)(bBase + (size_t)32 * 512);
    v16h b2 = *(const v16h*)(bBase + (size_t)64 * 512);
    v16h b3 = *(const v16h*)(bBase + (size_t)96 * 512);
    for (int kt = 0; kt < 15; ++kt) {
      const _Float16* bp = bBase + (size_t)(kt + 1) * NT_G * 512;
      v16h c0g = *(const v16h*)(bp + (size_t)0 * 512);
      v16h c1g = *(const v16h*)(bp + (size_t)32 * 512);
      v16h c2g = *(const v16h*)(bp + (size_t)64 * 512);
      v16h c3g = *(const v16h*)(bp + (size_t)96 * 512);
      v16h a = *(const v16h*)&hs[(mtile * 16 + kt) * 512 + lane * 16];
      acc[0] = WMMA_F16(a, b0, acc[0]);
      acc[1] = WMMA_F16(a, b1, acc[1]);
      acc[2] = WMMA_F16(a, b2, acc[2]);
      acc[3] = WMMA_F16(a, b3, acc[3]);
      b0 = c0g; b1 = c1g; b2 = c2g; b3 = c3g;
    }
    {
      v16h a = *(const v16h*)&hs[(mtile * 16 + 15) * 512 + lane * 16];
      acc[0] = WMMA_F16(a, b0, acc[0]);
      acc[1] = WMMA_F16(a, b1, acc[1]);
      acc[2] = WMMA_F16(a, b2, acc[2]);
      acc[3] = WMMA_F16(a, b3, acc[3]);
    }

    // ---- LN stats over the full 2048 gate vector ----
#pragma unroll
    for (int r = 0; r < 8; ++r) {
      float sv = acc[0][r] + acc[1][r] + acc[2][r] + acc[3][r];
      float sq = acc[0][r] * acc[0][r] + acc[1][r] * acc[1][r] +
                 acc[2][r] * acc[2][r] + acc[3][r] * acc[3][r];
#pragma unroll
      for (int msk = 8; msk >= 1; msk >>= 1) {
        sv += __shfl_xor(sv, msk, 32);
        sq += __shfl_xor(sq, msk, 32);
      }
      if (nlo == 0) {
        int m = mtile * 16 + r + 8 * hi;
        __hip_atomic_fetch_add(&gstats[(s * BATCH + m) * 2 + 0], sv,
                               __ATOMIC_RELAXED, __HIP_MEMORY_SCOPE_AGENT);
        __hip_atomic_fetch_add(&gstats[(s * BATCH + m) * 2 + 1], sq,
                               __ATOMIC_RELAXED, __HIP_MEMORY_SCOPE_AGENT);
      }
    }
    grid_sync(&bars[s * 3 + 0]);

    // ---- normalize, add gx(+bias), gates, c update, c stats ----
    float cnew[8], oreg[8];
#pragma unroll
    for (int r = 0; r < 8; ++r) {
      int m = mtile * 16 + r + 8 * hi;
      float mean = gstats[(s * BATCH + m) * 2 + 0] * (1.f / NG);
      float var = gstats[(s * BATCH + m) * 2 + 1] * (1.f / NG) - mean * mean;
      float rs = __frsqrt_rn(var + EPS);
      float gate[4];
#pragma unroll
      for (int g = 0; g < 4; ++g) {
        int n = (g * 32 + w * 2 + t) * 16 + nlo;
        gate[g] = (acc[g][r] - mean) * rs * lnhw[g] + lnhb[g] +
                  gx[(size_t)(s * BATCH + m) * NG + n];
      }
      float fg = fsig(gate[1] + 1.0f);  // forget bias
      float ig = fsig(gate[0]);
      float ug = tanhf(gate[3]);
      cnew[r] = fg * creg[r] + ig * ug;
      oreg[r] = gate[2];
      creg[r] = cnew[r];

      float sv = cnew[r], sq = cnew[r] * cnew[r];
#pragma unroll
      for (int msk = 8; msk >= 1; msk >>= 1) {
        sv += __shfl_xor(sv, msk, 32);
        sq += __shfl_xor(sq, msk, 32);
      }
      if (nlo == 0) {
        __hip_atomic_fetch_add(&cstats[(s * BATCH + m) * 2 + 0], sv,
                               __ATOMIC_RELAXED, __HIP_MEMORY_SCOPE_AGENT);
        __hip_atomic_fetch_add(&cstats[(s * BATCH + m) * 2 + 1], sq,
                               __ATOMIC_RELAXED, __HIP_MEMORY_SCOPE_AGENT);
      }
    }
    grid_sync(&bars[s * 3 + 1]);

    // ---- h = sig(o)*tanh(LN_H(c)); publish h (f32 out + f16 A-pack) ----
#pragma unroll
    for (int r = 0; r < 8; ++r) {
      int m = mtile * 16 + r + 8 * hi;
      float mc = cstats[(s * BATCH + m) * 2 + 0] * (1.f / HDIM);
      float vc = cstats[(s * BATCH + m) * 2 + 1] * (1.f / HDIM) - mc * mc;
      float rc = __frsqrt_rn(vc + EPS);
      float hv = fsig(oreg[r]) * tanhf((cnew[r] - mc) * rc * lncw + lncb);
      xout[(size_t)(s * BATCH + m) * HDIM + j] = hv;
      // repack into A-fragment layout for next step's WMMA
      int kt = j >> 5, kl = j & 31;
      int ahi = (kl >> 3) & 1;
      int e = (kl & 7) | ((kl >> 4) << 3);
      int alane = (m & 15) + (ahi << 4);
      hA[((size_t)(m >> 4) * 16 + kt) * 512 + alane * 16 + e] = (_Float16)hv;
      if (s == S_LEN - 1) {
        hfin[m * HDIM + j] = hv;
        cfin[m * HDIM + j] = cnew[r];
      }
    }
    grid_sync(&bars[s * 3 + 2]);
  }
}

// ---------------------------------------------------------------------------
extern "C" void kernel_launch(void* const* d_in, const int* in_sizes, int n_in,
                              void* d_out, int out_size, void* d_ws,
                              size_t ws_size, hipStream_t stream) {
  (void)in_sizes; (void)n_in; (void)out_size; (void)ws_size;
  const float* x_in = (const float*)d_in[0];
  const float* H0 = (const float*)d_in[1];
  const float* C0 = (const float*)d_in[2];
  const float* wx = (const float*)d_in[3];
  const float* wh = (const float*)d_in[4];
  const float* bias = (const float*)d_in[5];
  const float* lnA_w = (const float*)d_in[6];
  const float* lnA_b = (const float*)d_in[7];
  const float* lnB_w = (const float*)d_in[8];
  const float* lnB_b = (const float*)d_in[9];
  float* out = (float*)d_out;

  // ---- workspace carve ----
  char* ws = (char*)d_ws;
  _Float16* wxpk = (_Float16*)ws; ws += (size_t)2 * IDIM * NG * sizeof(_Float16);
  _Float16* whpk = (_Float16*)ws; ws += (size_t)2 * HDIM * NG * sizeof(_Float16);
  _Float16* Apk  = (_Float16*)ws; ws += (size_t)MROWS * IDIM * sizeof(_Float16);
  _Float16* hA   = (_Float16*)ws; ws += (size_t)2 * BATCH * HDIM * sizeof(_Float16);
  float* gx      = (float*)ws;    ws += (size_t)MROWS * NG * sizeof(float);
  float* x1      = (float*)ws;    ws += (size_t)MROWS * HDIM * sizeof(float);
  float* gstats  = (float*)ws;    ws += (size_t)S_LEN * BATCH * 2 * sizeof(float);
  float* cstats  = (float*)ws;    ws += (size_t)S_LEN * BATCH * 2 * sizeof(float);
  unsigned* bars = (unsigned*)ws; ws += (size_t)S_LEN * 3 * sizeof(unsigned);

  const size_t WSZ = (size_t)IDIM * NG;  // 1M halfs per layer weight

  // pack weights + initial h state (per launch; no cross-call state)
  for (int l = 0; l < 2; ++l) {
    pack_b_kernel<<<(IDIM * NG + 255) / 256, 256, 0, stream>>>(
        wx + (size_t)l * IDIM * NG, wxpk + (size_t)l * WSZ, IDIM, NG);
    pack_b_kernel<<<(HDIM * NG + 255) / 256, 256, 0, stream>>>(
        wh + (size_t)l * HDIM * NG, whpk + (size_t)l * WSZ, HDIM, NG);
    pack_a_kernel<<<(BATCH * HDIM + 255) / 256, 256, 0, stream>>>(
        H0 + (size_t)l * BATCH * HDIM, hA + (size_t)l * BATCH * HDIM, BATCH,
        HDIM);
  }

  float* xfin = out;
  float* hfin = out + (size_t)MROWS * HDIM;
  float* cfin = hfin + (size_t)2 * BATCH * HDIM;

  for (int l = 0; l < 2; ++l) {
    (void)hipMemsetAsync(gstats, 0, (size_t)S_LEN * BATCH * 2 * sizeof(float), stream);
    (void)hipMemsetAsync(cstats, 0, (size_t)S_LEN * BATCH * 2 * sizeof(float), stream);
    (void)hipMemsetAsync(bars, 0, (size_t)S_LEN * 3 * sizeof(unsigned), stream);

    const float* xsrc = (l == 0) ? x_in : x1;
    float* xdst = (l == 0) ? x1 : xfin;

    pack_a_kernel<<<(MROWS * IDIM + 255) / 256, 256, 0, stream>>>(
        xsrc, Apk, MROWS, IDIM);
    gemm_wmma_kernel<<<dim3(NT_G / 4, MROWS / 64), 128, 0, stream>>>(
        Apk, wxpk + (size_t)l * WSZ, gx);
    ln_rows_kernel<<<MROWS / 4, 128, 0, stream>>>(
        gx, lnA_w + (size_t)(2 * l) * NG, lnA_b + (size_t)(2 * l) * NG,
        bias + (size_t)l * NG);
    lstm_scan_kernel<<<NWG_REC, REC_THREADS, 0, stream>>>(
        gx, whpk + (size_t)l * WSZ, hA + (size_t)l * BATCH * HDIM,
        C0 + (size_t)l * BATCH * HDIM, lnA_w + (size_t)(2 * l + 1) * NG,
        lnA_b + (size_t)(2 * l + 1) * NG, lnB_w + (size_t)l * HDIM,
        lnB_b + (size_t)l * HDIM, xdst, hfin + (size_t)l * BATCH * HDIM,
        cfin + (size_t)l * BATCH * HDIM, gstats, cstats, bars);
  }
}
